// CSNA_42365557408054
// MI455X (gfx1250) — matrix operand order repaired and checked by reading, not compiled, not run on hardware
//
#include <hip/hip_runtime.h>
#include <hip/hip_bf16.h>

typedef __attribute__((ext_vector_type(16))) __bf16 v16bf;
typedef __attribute__((ext_vector_type(2)))  __bf16 v2bf;
typedef __attribute__((ext_vector_type(8)))  float  v8f;
typedef __attribute__((ext_vector_type(2)))  float  v2f;

#define HID   128
#define INC   500
#define OUTC  40
#define BN_EPS 1e-5f

union FragBF { v16bf v; unsigned int u[8]; };

__device__ __forceinline__ unsigned int pack_bf16(float x, float y) {
    v2f f = {x, y};
    v2bf b = __builtin_convertvector(f, v2bf);   // v_cvt_pk_bf16_f32 (RNE)
    unsigned int u;
    __builtin_memcpy(&u, &b, 4);
    return u;
}

// ---------------------------------------------------------------------------
// Specialized hidden-layer GEMM: C[M,128] = A[M,128] @ W[128,128]^T (+bias).
// W is staged ONCE per workgroup into LDS in bf16 WMMA-B fragment layout
// (32KB). Each wave then computes a 16-row strip: all 4 A fragments are
// preloaded from global (issued before any WMMA -> latency overlapped), and
// the 4x8 WMMA grid reads B fragments with ds_load_b128 only.
// ---------------------------------------------------------------------------
__global__ void __launch_bounds__(256)
gemm128_bf16_wmma_lds(const float* __restrict__ A, const float* __restrict__ W,
                      float* __restrict__ C, int M, const float* __restrict__ bias)
{
    __shared__ unsigned int Blds[8192];          // [pair=nt*4+kb][h=2][lane=32][4 dw]
    const int tid  = threadIdx.x;
    const int wave = tid >> 5;
    const int lane = tid & 31;
    const int half = lane >> 4;
    const int l15  = lane & 15;
    const int kloB = half << 4;                  // B frag: K{0..15} / K{16..31}

    // --- cooperative W stage: wave w fills pairs w*4 .. w*4+3 ---
#pragma unroll
    for (int i = 0; i < 4; ++i) {
        const int p  = wave * 4 + i;
        const int nt = p >> 2, kb = p & 3;
        const float* Wr = W + (nt * 16 + l15) * HID + kb * 32 + kloB;  // 64B aligned
        float4 w0 = ((const float4*)Wr)[0];
        float4 w1 = ((const float4*)Wr)[1];
        float4 w2 = ((const float4*)Wr)[2];
        float4 w3 = ((const float4*)Wr)[3];
        uint4 d0 = { pack_bf16(w0.x, w0.y), pack_bf16(w0.z, w0.w),
                     pack_bf16(w1.x, w1.y), pack_bf16(w1.z, w1.w) };
        uint4 d1 = { pack_bf16(w2.x, w2.y), pack_bf16(w2.z, w2.w),
                     pack_bf16(w3.x, w3.y), pack_bf16(w3.z, w3.w) };
        *(uint4*)&Blds[(p * 2 + 0) * 128 + lane * 4] = d0;
        *(uint4*)&Blds[(p * 2 + 1) * 128 + lane * 4] = d1;
    }
    __syncthreads();                             // all waves reach this

    const int strip = (blockIdx.x << 3) + wave;
    if (strip >= (M >> 4)) return;
    const int tm   = strip << 4;
    const int kloA = half << 4;                  // A frag: K{0..15} / K{16..31}
    const float* Arow = A + (long)(tm + l15) * HID;

    // --- preload all 4 A fragments (all global loads issued up front) ---
    FragBF a[4];
#pragma unroll
    for (int kb = 0; kb < 4; ++kb) {
        const float* p0 = Arow + kb * 32 + kloA;       // 32B aligned
        float4 x0 = ((const float4*)p0)[0];
        float4 x1 = ((const float4*)p0)[1];
        float4 y0 = ((const float4*)(p0 + 8))[0];
        float4 y1 = ((const float4*)(p0 + 8))[1];
        a[kb].u[0] = pack_bf16(x0.x, x0.y); a[kb].u[1] = pack_bf16(x0.z, x0.w);
        a[kb].u[2] = pack_bf16(x1.x, x1.y); a[kb].u[3] = pack_bf16(x1.z, x1.w);
        a[kb].u[4] = pack_bf16(y0.x, y0.y); a[kb].u[5] = pack_bf16(y0.z, y0.w);
        a[kb].u[6] = pack_bf16(y1.x, y1.y); a[kb].u[7] = pack_bf16(y1.z, y1.w);
    }

    v8f acc[8];
#pragma unroll
    for (int nt = 0; nt < 8; ++nt) acc[nt] = v8f{0.f,0.f,0.f,0.f,0.f,0.f,0.f,0.f};

#pragma unroll
    for (int kb = 0; kb < 4; ++kb) {
#pragma unroll
        for (int nt = 0; nt < 8; ++nt) {
            const int p = nt * 4 + kb;
            uint4 lo = *(const uint4*)&Blds[(p * 2 + 0) * 128 + lane * 4];
            uint4 hi = *(const uint4*)&Blds[(p * 2 + 1) * 128 + lane * 4];
            FragBF b;
            b.u[0] = lo.x; b.u[1] = lo.y; b.u[2] = lo.z; b.u[3] = lo.w;
            b.u[4] = hi.x; b.u[5] = hi.y; b.u[6] = hi.z; b.u[7] = hi.w;
            acc[nt] = __builtin_amdgcn_wmma_f32_16x16x32_bf16(
                false, a[kb].v, false, b.v, (short)0, acc[nt], false, false);
        }
    }

#pragma unroll
    for (int nt = 0; nt < 8; ++nt) {
        const int col = nt * 16 + l15;
        const float bia = bias ? bias[col] : 0.f;
#pragma unroll
        for (int r = 0; r < 8; ++r) {
            const int row = tm + r + (half << 3);
            C[(long)row * HID + col] = acc[nt][r] + bia;
        }
    }
}

// ---------------------------------------------------------------------------
// Generic strip GEMM (MLP K=500 and classifier Nout=40).
// GUARD=false compiles out all column-validity selects.
// ---------------------------------------------------------------------------
template <int NT, bool GUARD>
__global__ void __launch_bounds__(256)
gemm_bf16_wmma(const float* __restrict__ A, int lda, int M, int K,
               const float* __restrict__ W, int ldw, int Nout,
               float* __restrict__ C, int ldc,
               const float* __restrict__ bias,
               const float* __restrict__ bn_g, const float* __restrict__ bn_b,
               const float* __restrict__ bn_m, const float* __restrict__ bn_v,
               int relu)
{
    const int strip = (blockIdx.x << 3) + (threadIdx.x >> 5);
    if (strip >= (M >> 4)) return;

    const int tm   = strip << 4;
    const int lane = threadIdx.x & 31;
    const int half = lane >> 4;
    const int l15  = lane & 15;
    const int kloA = half << 4;
    const int kloB = half << 4;

    const float* Arow = A + (long)(tm + l15) * lda;

    const float* Wrow[NT];
    bool wvalid[NT];
#pragma unroll
    for (int nt = 0; nt < NT; ++nt) {
        int wr = nt * 16 + l15;
        wvalid[nt] = !GUARD || (wr < Nout);
        if (GUARD && wr >= Nout) wr = Nout - 1;  // clamp: address always valid
        Wrow[nt] = W + (long)wr * ldw;
    }

    v8f acc[NT];
#pragma unroll
    for (int nt = 0; nt < NT; ++nt) acc[nt] = v8f{0.f,0.f,0.f,0.f,0.f,0.f,0.f,0.f};

    const int KTfull = K >> 5;
    for (int kb = 0; kb < KTfull; ++kb) {
        const int kbase = kb << 5;
        FragBF a;
#pragma unroll
        for (int r = 0; r < 8; ++r) {
            const int koffA = r << 1;
            float2 av = *(const float2*)(Arow + kbase + kloA + koffA);
            a.u[r] = pack_bf16(av.x, av.y);
        }
#pragma unroll
        for (int nt = 0; nt < NT; ++nt) {
            FragBF b;
#pragma unroll
            for (int r = 0; r < 8; ++r) {
                float2 bv = *(const float2*)(Wrow[nt] + kbase + kloB + (r << 1));
                float b0 = (!GUARD || wvalid[nt]) ? bv.x : 0.f;
                float b1 = (!GUARD || wvalid[nt]) ? bv.y : 0.f;
                b.u[r] = pack_bf16(b0, b1);
            }
            acc[nt] = __builtin_amdgcn_wmma_f32_16x16x32_bf16(
                false, a.v, false, b.v, (short)0, acc[nt], false, false);
        }
    }

    if (K & 31) {                        // branchless tail chunk (K=500 path)
        const int kbase = KTfull << 5;
        FragBF a;
#pragma unroll
        for (int r = 0; r < 8; ++r) {
            const int koffA = r << 1;
            const int k0 = kbase + kloA + koffA;
            float a0 = Arow[min(k0,     K - 1)];
            float a1 = Arow[min(k0 + 1, K - 1)];
            a0 = (k0     < K) ? a0 : 0.f;
            a1 = (k0 + 1 < K) ? a1 : 0.f;
            a.u[r] = pack_bf16(a0, a1);
        }
#pragma unroll
        for (int nt = 0; nt < NT; ++nt) {
            FragBF b;
#pragma unroll
            for (int r = 0; r < 8; ++r) {
                const int k0 = kbase + kloB + (r << 1);
                float b0 = Wrow[nt][min(k0,     K - 1)];
                float b1 = Wrow[nt][min(k0 + 1, K - 1)];
                b0 = (wvalid[nt] && k0     < K) ? b0 : 0.f;
                b1 = (wvalid[nt] && k0 + 1 < K) ? b1 : 0.f;
                b.u[r] = pack_bf16(b0, b1);
            }
            acc[nt] = __builtin_amdgcn_wmma_f32_16x16x32_bf16(
                false, a.v, false, b.v, (short)0, acc[nt], false, false);
        }
    }

#pragma unroll
    for (int nt = 0; nt < NT; ++nt) {
        const int col = nt * 16 + l15;
        if (!GUARD || col < Nout) {
            float bia = bias ? bias[col] : 0.f;
            float sc = 1.f, sh = 0.f;
            if (bn_g) {
                sc = bn_g[col] * rsqrtf(bn_v[col] + BN_EPS);
                sh = bn_b[col] - bn_m[col] * sc;
            }
#pragma unroll
            for (int r = 0; r < 8; ++r) {
                const int row = tm + r + (half << 3);
                float v = acc[nt][r] + bia;
                v = v * sc + sh;
                if (relu) v = fmaxf(v, 0.f);
                C[(long)row * ldc + col] = v;
            }
        }
    }
}

// ---------------------------------------------------------------------------
// Edge pass 1: s_e = sigmoid(-||xg[row]-xg[col]||); segment max via int atomicMax
// ---------------------------------------------------------------------------
__global__ void __launch_bounds__(256)
edge_pass1(const int* __restrict__ row, const int* __restrict__ col, int E,
           const float* __restrict__ xg, float* __restrict__ sbuf,
           float* mcon, float* mdis)
{
    const int e = (blockIdx.x << 3) + (threadIdx.x >> 5);
    if (e >= E) return;
    const int lane = threadIdx.x & 31;
    const int r = row[e], c = col[e];
    float4 a = ((const float4*)(xg + (long)r * HID))[lane];
    float4 b = ((const float4*)(xg + (long)c * HID))[lane];
    float dx = a.x - b.x, dy = a.y - b.y, dz = a.z - b.z, dw = a.w - b.w;
    float sq = dx * dx + dy * dy + dz * dz + dw * dw;
#pragma unroll
    for (int off = 16; off > 0; off >>= 1) sq += __shfl_xor(sq, off, 32);
    if (lane == 0) {
        float g = (sq > 0.f) ? sqrtf(sq) : 0.f;
        float s = 1.f / (1.f + expf(g));       // sigmoid(-g), TAU=1
        sbuf[e] = s;
        atomicMax((int*)(mcon + r), __float_as_int(s));         // s in (0,0.5] > 0
        atomicMax((int*)(mdis + r), __float_as_int(1.f - s));   // in [0.5,1) > 0
    }
}

// ---------------------------------------------------------------------------
// Edge pass 2: softmax denominators (thread per edge)
// ---------------------------------------------------------------------------
__global__ void __launch_bounds__(256)
edge_pass2(const int* __restrict__ row, int E, const float* __restrict__ sbuf,
           const float* __restrict__ mcon, const float* __restrict__ mdis,
           float* zcon, float* zdis)
{
    const int e = blockIdx.x * 256 + threadIdx.x;
    if (e >= E) return;
    const int r = row[e];
    const float s = sbuf[e];
    atomicAdd(zcon + r, expf(s - mcon[r]));
    atomicAdd(zdis + r, expf((1.f - s) - mdis[r]));
}

// ---------------------------------------------------------------------------
// Edge pass 3: weighted scatter-add of x_con[col] / x_dis[col] into row segments
// ---------------------------------------------------------------------------
__global__ void __launch_bounds__(256)
edge_pass3(const int* __restrict__ row, const int* __restrict__ col, int E,
           const float* __restrict__ sbuf,
           const float* __restrict__ mcon, const float* __restrict__ mdis,
           const float* __restrict__ zcon, const float* __restrict__ zdis,
           const float* __restrict__ xcon, const float* __restrict__ xdis,
           float* ocon, float* odis)
{
    const int e = (blockIdx.x << 3) + (threadIdx.x >> 5);
    if (e >= E) return;
    const int lane = threadIdx.x & 31;
    const int r = row[e], c = col[e];
    const float s = sbuf[e];
    const float wc = expf(s - mcon[r]) / (zcon[r] + 1e-16f);
    const float wd = expf((1.f - s) - mdis[r]) / (zdis[r] + 1e-16f);
    float4 xc = ((const float4*)(xcon + (long)c * HID))[lane];
    float4 xd = ((const float4*)(xdis + (long)c * HID))[lane];
    float* oc = ocon + (long)r * HID + lane * 4;
    float* od = odis + (long)r * HID + lane * 4;
    atomicAdd(oc + 0, wc * xc.x); atomicAdd(oc + 1, wc * xc.y);
    atomicAdd(oc + 2, wc * xc.z); atomicAdd(oc + 3, wc * xc.w);
    atomicAdd(od + 0, wd * xd.x); atomicAdd(od + 1, wd * xd.y);
    atomicAdd(od + 2, wd * xd.z); atomicAdd(od + 3, wd * xd.w);
}

// ---------------------------------------------------------------------------
// Combine: 3-way gate softmax, optional BN+ReLU (layer 0), residual.
// One wave per node. NOTE: odis may alias hout (read happens before write).
// ---------------------------------------------------------------------------
__global__ void __launch_bounds__(256)
combine_kernel(int N, const float* ocon, const float* odis, const float* oself,
               const float* hin,
               const float* __restrict__ Wgate, const float* __restrict__ bgate,
               const float* __restrict__ bn_g, const float* __restrict__ bn_b,
               const float* __restrict__ bn_m, const float* __restrict__ bn_v,
               int do_bnrelu, float* hout)
{
    const int i = (blockIdx.x << 3) + (threadIdx.x >> 5);
    if (i >= N) return;
    const int lane = threadIdx.x & 31;
    const int k = lane * 4;
    float4 oc4 = ((const float4*)(ocon  + (long)i * HID))[lane];
    float4 od4 = ((const float4*)(odis  + (long)i * HID))[lane];
    float4 os4 = ((const float4*)(oself + (long)i * HID))[lane];
    float cc[4] = {oc4.x, oc4.y, oc4.z, oc4.w};
    float dd[4] = {od4.x, od4.y, od4.z, od4.w};
    float ss[4] = {os4.x, os4.y, os4.z, os4.w};

    float l0 = 0.f, l1 = 0.f, l2 = 0.f;
#pragma unroll
    for (int t = 0; t < 4; ++t) {
        const int kc = k + t;
        l0 += Wgate[kc] * cc[t] + Wgate[128 + kc] * dd[t] + Wgate[256 + kc] * ss[t];
        l1 += Wgate[384 + kc] * cc[t] + Wgate[512 + kc] * dd[t] + Wgate[640 + kc] * ss[t];
        l2 += Wgate[768 + kc] * cc[t] + Wgate[896 + kc] * dd[t] + Wgate[1024 + kc] * ss[t];
    }
#pragma unroll
    for (int off = 16; off > 0; off >>= 1) {
        l0 += __shfl_xor(l0, off, 32);
        l1 += __shfl_xor(l1, off, 32);
        l2 += __shfl_xor(l2, off, 32);
    }
    l0 += bgate[0]; l1 += bgate[1]; l2 += bgate[2];
    const float mx = fmaxf(l0, fmaxf(l1, l2));
    const float e0 = expf(l0 - mx), e1 = expf(l1 - mx), e2 = expf(l2 - mx);
    const float inv = 1.f / (e0 + e1 + e2);
    const float g0 = e0 * inv, g1 = e1 * inv, g2 = e2 * inv;

    float4 h4 = ((const float4*)(hin + (long)i * HID))[lane];
    float hh[4] = {h4.x, h4.y, h4.z, h4.w};
    float out[4];
#pragma unroll
    for (int t = 0; t < 4; ++t) {
        float v = g0 * cc[t] + g1 * dd[t] + g2 * ss[t];
        if (do_bnrelu) {
            const int ch = k + t;
            v = (v - bn_m[ch]) * (bn_g[ch] * rsqrtf(bn_v[ch] + BN_EPS)) + bn_b[ch];
            v = fmaxf(v, 0.f);
        }
        out[t] = v + hh[t];
    }
    float4 o4 = {out[0], out[1], out[2], out[3]};
    ((float4*)(hout + (long)i * HID))[lane] = o4;
}

// ---------------------------------------------------------------------------
extern "C" void kernel_launch(void* const* d_in, const int* in_sizes, int n_in,
                              void* d_out, int out_size, void* d_ws, size_t ws_size,
                              hipStream_t stream)
{
    const int N = in_sizes[0] / INC;          // 100000
    const int E = in_sizes[1] / 2;            // 1700000

    // JAX tree-flatten order: dict keys sorted ('W' < 'b'), lists in order.
    const float* x     = (const float*)d_in[0];
    const int*   ei    = (const int*)d_in[1];
    const int*   rowp  = ei;
    const int*   colp  = ei + E;
    const float* bn0_b = (const float*)d_in[2];
    const float* bn0_g = (const float*)d_in[3];
    const float* bn0_m = (const float*)d_in[4];
    const float* bn0_v = (const float*)d_in[5];
    const float* cls_b = (const float*)d_in[6];
    const float* cls_w = (const float*)d_in[7];
    struct ConvP { const float *Wcon, *Wdis, *Wg, *Wgate, *Wself, *bgate, *bself; } cp[2];
    for (int l = 0; l < 2; ++l) {
        const int b = 8 + l * 7;
        cp[l].Wcon  = (const float*)d_in[b + 0];
        cp[l].Wdis  = (const float*)d_in[b + 1];
        cp[l].Wg    = (const float*)d_in[b + 2];
        cp[l].Wgate = (const float*)d_in[b + 3];
        cp[l].Wself = (const float*)d_in[b + 4];
        cp[l].bgate = (const float*)d_in[b + 5];
        cp[l].bself = (const float*)d_in[b + 6];
    }
    const float* mlp_b = (const float*)d_in[22];
    const float* mbn_b = (const float*)d_in[23];
    const float* mbn_g = (const float*)d_in[24];
    const float* mbn_m = (const float*)d_in[25];
    const float* mbn_v = (const float*)d_in[26];
    const float* mlp_w = (const float*)d_in[27];
    float* out = (float*)d_out;

    // Workspace partition (floats). Node matrices are 51MB each -> L2 resident.
    float* ws = (float*)d_ws;
    const size_t nh = (size_t)N * HID;
    float* hA    = ws; ws += nh;
    float* hB    = ws; ws += nh;
    float* xg    = ws; ws += nh;          // reused as ocon after pass 1
    float* xcon  = ws; ws += nh;
    float* xdis  = ws; ws += nh;
    float* oself = ws; ws += nh;
    float* sbuf  = ws; ws += E;
    float* mcon  = ws; ws += N;
    float* mdis  = ws; ws += N;
    float* zcon  = ws; ws += N;
    float* zdis  = ws; ws += N;

    const dim3 blk(256);
    const int stripBlocks    = ((N / 16) + 7) / 8;    // one wave per 16-row strip
    const int edgeWaveBlocks = (E + 7) / 8;
    const int edgeThrBlocks  = (E + 255) / 256;
    const int nodeWaveBlocks = (N + 7) / 8;

    // Input MLP: h0 = relu(BN(x @ mlp_w^T + mlp_b))   [K=500: 15 chunks + tail]
    gemm_bf16_wmma<8, false><<<stripBlocks, blk, 0, stream>>>(
        x, INC, N, INC, mlp_w, INC, HID, hA, HID,
        mlp_b, mbn_g, mbn_b, mbn_m, mbn_v, 1);

    for (int l = 0; l < 2; ++l) {
        const float* hin  = l ? hB : hA;
        float*       hout = l ? hA : hB;
        float*       ocon = xg;           // xg dead after pass 1
        float*       odis = hout;         // hout dead until combine writes it

        gemm128_bf16_wmma_lds<<<stripBlocks, blk, 0, stream>>>(hin, cp[l].Wg,   xg,    N, nullptr);
        gemm128_bf16_wmma_lds<<<stripBlocks, blk, 0, stream>>>(hin, cp[l].Wcon, xcon,  N, nullptr);
        gemm128_bf16_wmma_lds<<<stripBlocks, blk, 0, stream>>>(hin, cp[l].Wdis, xdis,  N, nullptr);
        gemm128_bf16_wmma_lds<<<stripBlocks, blk, 0, stream>>>(hin, cp[l].Wself, oself, N, cp[l].bself);

        hipMemsetAsync(mcon, 0, (size_t)N * 4, stream);
        hipMemsetAsync(mdis, 0, (size_t)N * 4, stream);
        hipMemsetAsync(zcon, 0, (size_t)N * 4, stream);
        hipMemsetAsync(zdis, 0, (size_t)N * 4, stream);

        edge_pass1<<<edgeWaveBlocks, blk, 0, stream>>>(rowp, colp, E, xg, sbuf, mcon, mdis);
        edge_pass2<<<edgeThrBlocks, blk, 0, stream>>>(rowp, E, sbuf, mcon, mdis, zcon, zdis);

        hipMemsetAsync(ocon, 0, nh * 4, stream);
        hipMemsetAsync(odis, 0, nh * 4, stream);

        edge_pass3<<<edgeWaveBlocks, blk, 0, stream>>>(rowp, colp, E, sbuf,
            mcon, mdis, zcon, zdis, xcon, xdis, ocon, odis);

        combine_kernel<<<nodeWaveBlocks, blk, 0, stream>>>(N, ocon, odis, oself, hin,
            cp[l].Wgate, cp[l].bgate, bn0_g, bn0_b, bn0_m, bn0_v, (l == 0) ? 1 : 0, hout);
    }

    // Classifier: out = h @ cls_w^T + cls_b  (40 cols -> NT=3, guarded)
    gemm_bf16_wmma<3, true><<<stripBlocks, blk, 0, stream>>>(
        hA, HID, N, HID, cls_w, HID, OUTC, out, OUTC,
        cls_b, nullptr, nullptr, nullptr, nullptr, 0);
}
